// HRAMModel_19310172963570
// MI455X (gfx1250) — matrix operand order repaired and checked by reading, not compiled, not run on hardware
//
#include <hip/hip_runtime.h>
#include <hip/hip_bf16.h>
#include <math.h>

// Problem dims (fixed by the reference)
#define TDIM 128
#define BDIM 64
#define SDIM 16
#define EDIM 512
#define HDIM 512
#define HH   256
#define G4   1024                       // 4*HH
#define BT   (BDIM*TDIM)                // 8192
#define ROWS (BDIM*TDIM*SDIM)           // 131072

typedef __attribute__((ext_vector_type(16))) _Float16 v16h;
typedef __attribute__((ext_vector_type(8)))  _Float16 v8h;
typedef __attribute__((ext_vector_type(8)))  float    v8f;

__device__ __forceinline__ float sigf(float x){ return 1.0f/(1.0f+__expf(-x)); }

// ---- WMMA fragment helpers (CDNA5 16x16x32 f16, wave32) -------------------
// A fragment (16x32, f16): lane&15 = M row; lanes 0-15 K base 0, lanes 16-31 K base 8;
// v16h elements j=0..7 -> K = kb+j ; j=8..15 -> K = kb+16+(j-8)
__device__ __forceinline__ v16h frag_a(const _Float16* A, int lda, unsigned lane){
  unsigned m  = lane & 15u;
  unsigned kb = (lane & 16u) ? 8u : 0u;
  const _Float16* p = A + (size_t)m*lda + kb;
  v8h lo = *(const v8h*)(p);
  v8h hi = *(const v8h*)(p + 16);
  v16h f;
#pragma unroll
  for (int j=0;j<8;++j){ f[j]=lo[j]; f[j+8]=hi[j]; }
  return f;
}
// B fragment (32x16, f16, k-major storage Bt[k][n]): lane&15 = N col; K order same as frag_a
__device__ __forceinline__ v16h frag_b(const _Float16* Bt, int ldb, unsigned lane){
  unsigned n  = lane & 15u;
  unsigned kb = (lane & 16u) ? 8u : 0u;
  v16h f;
#pragma unroll
  for (int j=0;j<8;++j){ f[j]=Bt[(size_t)(kb+j)*ldb + n]; f[j+8]=Bt[(size_t)(kb+16+j)*ldb + n]; }
  return f;
}
__device__ __forceinline__ v8f wmma_f16(v16h a, v16h b, v8f c){
  return __builtin_amdgcn_wmma_f32_16x16x32_f16(false, a, false, b, (short)0, c, false, false);
}

// ---- small utility kernels ------------------------------------------------
__global__ void k_cvt(const float* __restrict__ s, _Float16* __restrict__ d, size_t n){
  size_t i = (size_t)blockIdx.x*blockDim.x + threadIdx.x;
  if (i < n) d[i] = (_Float16)s[i];
}
// src [N][K] f32 -> dst [K][N] f16  (k-major for WMMA B operand)
__global__ void k_convT(const float* __restrict__ s, _Float16* __restrict__ d, int N, int K){
  int i = blockIdx.x*blockDim.x + threadIdx.x;
  if (i < N*K){ int n = i / K, k = i % K; d[(size_t)k*N + n] = (_Float16)s[i]; }
}
__global__ void k_bias_sum(const float* __restrict__ a, const float* __restrict__ b,
                           float* __restrict__ c, int n){
  int i = blockIdx.x*blockDim.x + threadIdx.x;
  if (i < n) c[i] = a[i] + b[i];
}

// ---- generic f16 GEMM: C = act(A[M][K] * Bt[K][N] + bias) -----------------
// block = 256 thr (8 waves); block tile 128(M) x 64(N); wave = one 16-row M strip.
// OUTMODE: 0 = f16 row-major, 1 = f32 row-major, 2 = f32 LSTM-xg remap
template<int OUTMODE>
__global__ __launch_bounds__(256) void k_gemm(const _Float16* __restrict__ A,
        const _Float16* __restrict__ Bt, const float* __restrict__ bias,
        void* __restrict__ C, int M, int N, int K, int act)
{
  __shared__ alignas(32) _Float16 as[128][32];
  const int tid = threadIdx.x;
  const unsigned lane = tid & 31u;
  const int w  = tid >> 5;
  const int m0 = blockIdx.y * 128;
  const int n0 = blockIdx.x * 64;
  const v8f vz = {};
  v8f acc[4] = {vz, vz, vz, vz};
  const int kcn = K >> 5;
#pragma unroll 1
  for (int kc = 0; kc < kcn; ++kc){
    { // stage A chunk 128x32 (f16), one 32B vector per thread
      int r = tid >> 1, off = (tid & 1) * 16;
      *(v16h*)&as[r][off] = *(const v16h*)(A + (size_t)(m0 + r)*K + kc*32 + off);
      if (kc + 1 < kcn)  // gfx1250 global_prefetch_b8 path
        __builtin_prefetch(A + (size_t)(m0 + r)*K + (kc+1)*32 + off, 0, 1);
    }
    __syncthreads();
    v16h af = frag_a(&as[w*16][0], 32, lane);
    const _Float16* Bp = Bt + (size_t)(kc*32)*N + n0;
#pragma unroll
    for (int ns=0; ns<4; ++ns){
      v16h bf = frag_b(Bp + ns*16, N, lane);
      acc[ns] = wmma_f16(af, bf, acc[ns]);
    }
    __syncthreads();
  }
  const int msel = (lane & 16u) ? 8 : 0;
#pragma unroll
  for (int ns=0; ns<4; ++ns){
    int n = n0 + ns*16 + (lane & 15);
    float bv = bias ? bias[n] : 0.0f;
#pragma unroll
    for (int r=0; r<8; ++r){
      int m = m0 + w*16 + r + msel;
      float v = acc[ns][r] + bv;
      if (act == 1) v = fmaxf(v, 0.0f);
      else if (act == 2) v = tanhf(v);
      if (OUTMODE == 0)      ((_Float16*)C)[(size_t)m*N + n] = (_Float16)v;
      else if (OUTMODE == 1) ((float*)C)[(size_t)m*N + n] = v;
      else { int b = m >> 7, t = m & 127;   // m = b*T+t -> xg[t][b][n]
             ((float*)C)[((size_t)t*BDIM + b)*G4 + n] = v; }
    }
  }
}

// ---- bidirectional LSTM scan (one layer, both dirs via blockIdx.y) --------
// Recurrence independent per batch row -> block owns 16 batch rows, loops t.
// Per step: g[16][1024] = xg[t] + h16 @ WhhT via WMMA; cell update in LDS.
// Register-pressure bounded: wave's 128 gate cols done in 2 passes of 64
// (acc[4] live, K-loop not unrolled -> no scratch spills).
__global__ __launch_bounds__(256) void k_lstm(const float* __restrict__ xg,      // [2][T][B][1024]
        const _Float16* __restrict__ whhT,                                        // [2][256][1024]
        const int* __restrict__ seq_lens,
        float* __restrict__ out)                                                  // [B][T][512]
{
  __shared__ alignas(32) _Float16 h16[16][256];
  __shared__ alignas(32) _Float16 gbuf[16][1024];
  const int tid = threadIdx.x;
  const unsigned lane = tid & 31u;
  const int w  = tid >> 5;
  const int b0 = blockIdx.x * 16;
  const int d  = blockIdx.y;
  for (int i = tid; i < 16*256; i += 256) (&h16[0][0])[i] = (_Float16)0.0f;
  const int rr = tid >> 4;             // 0..15 batch row for cell update
  const int hb = (tid & 15) * 16;      // hidden base, 16 units/thread
  const int len = seq_lens[b0 + rr];
  float creg[16];
#pragma unroll
  for (int j=0;j<16;++j) creg[j] = 0.0f;
  __syncthreads();
  const _Float16* whhTd = whhT + (size_t)d * 256 * 1024;
  const int msel = (lane & 16u) ? 8 : 0;
#pragma unroll 1
  for (int tt = 0; tt < TDIM; ++tt){
    const int t = d ? (TDIM - 1 - tt) : tt;
    const size_t xrow = ((size_t)d*TDIM + t) * BDIM + b0;
    // ---- gate GEMM: wave w covers gate cols [w*128, w*128+128), 2 passes of 64
#pragma unroll 1
    for (int half = 0; half < 2; ++half){
      const int nb = w*128 + half*64;
      v8f acc[4];
#pragma unroll
      for (int ns=0; ns<4; ++ns){
        int n = nb + ns*16 + (lane & 15);
#pragma unroll
        for (int r=0; r<8; ++r) acc[ns][r] = xg[(xrow + r + msel)*G4 + n];
      }
#pragma unroll 1
      for (int kc=0; kc<8; ++kc){
        v16h af = frag_a(&h16[0][kc*32], 256, lane);
        const _Float16* Bp = whhTd + (size_t)(kc*32)*1024 + nb;
#pragma unroll
        for (int ns=0; ns<4; ++ns){
          v16h bf = frag_b(Bp + ns*16, 1024, lane);
          acc[ns] = wmma_f16(af, bf, acc[ns]);
        }
      }
#pragma unroll
      for (int ns=0; ns<4; ++ns){
        int n = nb + ns*16 + (lane & 15);
#pragma unroll
        for (int r=0; r<8; ++r) gbuf[r + msel][n] = (_Float16)acc[ns][r];
      }
    }
    __syncthreads();
    // ---- cell update, masked at t >= len (state frozen)
    const bool mk = (t < len);
#pragma unroll
    for (int j=0;j<16;++j){
      int hid = hb + j;
      float i_ = (float)gbuf[rr][hid];
      float f_ = (float)gbuf[rr][256+hid];
      float g_ = (float)gbuf[rr][512+hid];
      float o_ = (float)gbuf[rr][768+hid];
      float c2 = sigf(f_)*creg[j] + sigf(i_)*tanhf(g_);
      float h2 = sigf(o_)*tanhf(c2);
      if (mk){ creg[j] = c2; h16[rr][hid] = (_Float16)h2; }
      out[(((size_t)(b0+rr))*TDIM + t)*HDIM + d*HH + hid] = (float)h16[rr][hid];
    }
    __syncthreads();
  }
}

// ---- attention score reduce: sc[m] = sum_n tanh_tmp[m][n]*w2[n] ----------
__global__ void k_attn_reduce(const float* __restrict__ tt, const float* __restrict__ w2,
                              float* __restrict__ sc){
  int m = blockIdx.x*blockDim.x + threadIdx.x;
  if (m >= BT) return;
  float s = 0.0f;
  for (int n=0;n<HDIM;++n) s += tt[(size_t)m*HDIM + n] * w2[n];
  sc[m] = s;
}

// ---- causal softmax + ctx = a @ rnn, per batch element --------------------
__global__ __launch_bounds__(256) void k_ctx(const float* __restrict__ sc,
        const _Float16* __restrict__ rnn16, const int* __restrict__ seq_lens,
        float* __restrict__ ctx)
{
  __shared__ float scs[TDIM];
  __shared__ alignas(32) _Float16 aw[TDIM][TDIM];
  const int tid = threadIdx.x;
  const int b = blockIdx.x;
  if (tid < TDIM) scs[tid] = sc[b*TDIM + tid];
  __syncthreads();
  if (tid < TDIM){
    int q = tid;
    float mx = -1e30f;
    for (int s=0;s<=q;++s) mx = fmaxf(mx, scs[s]);
    float Z = 0.0f;
    for (int s=0;s<=q;++s) Z += __expf(scs[s]-mx);
    float inv = 1.0f/Z;
    for (int s=0;s<TDIM;++s)
      aw[q][s] = (s<=q) ? (_Float16)(__expf(scs[s]-mx)*inv) : (_Float16)0.0f;
  }
  __syncthreads();
  const unsigned lane = tid & 31u;
  const int w = tid >> 5;                 // 8 waves = 8 query strips of 16
  const int msel = (lane & 16u) ? 8 : 0;
  const int len = seq_lens[b];
  const _Float16* Bb = rnn16 + (size_t)b*TDIM*HDIM;
#pragma unroll 1
  for (int ns=0; ns<32; ++ns){            // 512 output cols
    v8f acc = {};
#pragma unroll
    for (int kc=0; kc<4; ++kc){
      v16h af = frag_a(&aw[w*16][kc*32], TDIM, lane);
      v16h bf = frag_b(Bb + (size_t)(kc*32)*HDIM + ns*16, HDIM, lane);
      acc = wmma_f16(af, bf, acc);
    }
    int n = ns*16 + (lane & 15);
#pragma unroll
    for (int r=0; r<8; ++r){
      int q = w*16 + r + msel;
      ctx[((size_t)b*TDIM + q)*HDIM + n] = (q < len) ? acc[r] : 0.0f;
    }
  }
}

// ---- embedding branch fused: relu(relu((ue*se)W1^T+b1)W2^T+b2) -> z0[:,0:64]
__global__ __launch_bounds__(128) void k_ebranch(
    const float* __restrict__ user_table, const float* __restrict__ art_table,
    const int* __restrict__ user_ids, const int* __restrict__ sample_ids,
    const _Float16* __restrict__ W1t, const float* __restrict__ b1,
    const _Float16* __restrict__ W2t, const float* __restrict__ b2,
    _Float16* __restrict__ z0)
{
  __shared__ alignas(32) _Float16 a0[64][128];
  __shared__ alignas(32) _Float16 h1[64][128];
  const int tid = threadIdx.x;
  const int row0 = blockIdx.x * 64;
  for (int idx = tid; idx < 64*128; idx += 128){
    int r = idx >> 7, k = idx & 127;
    int row = row0 + r;
    int b = row >> 11;                 // row / (T*S)
    int uid = user_ids[b];
    int aid = sample_ids[row];
    a0[r][k] = (_Float16)(user_table[(size_t)uid*128 + k] * art_table[(size_t)aid*128 + k]);
  }
  __syncthreads();
  const unsigned lane = tid & 31u;
  const int w = tid >> 5;
  const int msel = (lane & 16u) ? 8 : 0;
#pragma unroll 1
  for (int ns=0; ns<8; ++ns){          // GEMM1: 64x128, K=128
    v8f acc = {};
#pragma unroll
    for (int kc=0; kc<4; ++kc){
      v16h af = frag_a(&a0[w*16][kc*32], 128, lane);
      v16h bf = frag_b(W1t + (size_t)(kc*32)*128 + ns*16, 128, lane);
      acc = wmma_f16(af, bf, acc);
    }
    int n = ns*16 + (lane & 15);
    float bv = b1[n];
#pragma unroll
    for (int r=0; r<8; ++r)
      h1[w*16 + r + msel][n] = (_Float16)fmaxf(acc[r] + bv, 0.0f);
  }
  __syncthreads();
#pragma unroll 1
  for (int ns=0; ns<4; ++ns){          // GEMM2: 64x64, K=128
    v8f acc = {};
#pragma unroll
    for (int kc=0; kc<4; ++kc){
      v16h af = frag_a(&h1[w*16][kc*32], 128, lane);
      v16h bf = frag_b(W2t + (size_t)(kc*32)*64 + ns*16, 64, lane);
      acc = wmma_f16(af, bf, acc);
    }
    int n = ns*16 + (lane & 15);
    float bv = b2[n];
#pragma unroll
    for (int r=0; r<8; ++r){
      int m = row0 + w*16 + r + msel;
      z0[(size_t)m*128 + n] = (_Float16)fmaxf(acc[r] + bv, 0.0f);
    }
  }
}

// ---- rnn branch: (ctx broadcast * d2v) @ rnn_mlp_W^T + b -> z0[:,64:128] --
__global__ __launch_bounds__(256) void k_rbranch(
    const float* __restrict__ ctx, const float* __restrict__ d2v,
    const _Float16* __restrict__ Wt, const float* __restrict__ bias,
    _Float16* __restrict__ z0)
{
  __shared__ alignas(32) _Float16 a0[128][32];
  const int tid = threadIdx.x;
  const int row0 = blockIdx.x * 128;
  const unsigned lane = tid & 31u;
  const int w = tid >> 5;
  const int msel = (lane & 16u) ? 8 : 0;
  const v8f vz = {};
  v8f acc[4] = {vz, vz, vz, vz};
#pragma unroll 1
  for (int kc=0; kc<16; ++kc){         // K = 512 streamed in 32-chunks
    for (int idx = tid; idx < 128*32; idx += 256){
      int r = idx >> 5, kk = idx & 31;
      int row = row0 + r;
      int b = row >> 11;
      int t = (row >> 4) & 127;
      int k = kc*32 + kk;
      a0[r][kk] = (_Float16)(ctx[(((size_t)b)*TDIM + t)*HDIM + k] *
                             d2v[(size_t)row*HDIM + k]);
    }
    __syncthreads();
    v16h af = frag_a(&a0[w*16][0], 32, lane);
#pragma unroll
    for (int ns=0; ns<4; ++ns){
      v16h bf = frag_b(Wt + (size_t)(kc*32)*64 + ns*16, 64, lane);
      acc[ns] = wmma_f16(af, bf, acc[ns]);
    }
    __syncthreads();
  }
#pragma unroll
  for (int ns=0; ns<4; ++ns){
    int n = ns*16 + (lane & 15);
    float bv = bias[n];
#pragma unroll
    for (int r=0; r<8; ++r){
      int m = row0 + w*16 + r + msel;
      z0[(size_t)m*128 + 64 + n] = (_Float16)(acc[ns][r] + bv);  // no relu (matches ref)
    }
  }
}

// ---- final: sigmoid(z3 . W4 + b4) ----------------------------------------
__global__ void k_final(const _Float16* __restrict__ z3, const float* __restrict__ W4,
                        const float* __restrict__ b4, float* __restrict__ out){
  int m = blockIdx.x*blockDim.x + threadIdx.x;
  if (m >= ROWS) return;
  float s = b4[0];
  for (int j=0;j<64;++j) s += (float)z3[(size_t)m*64 + j] * W4[j];
  out[m] = 1.0f/(1.0f+__expf(-s));
}

// ===========================================================================
extern "C" void kernel_launch(void* const* d_in, const int* in_sizes, int n_in,
                              void* d_out, int out_size, void* d_ws, size_t ws_size,
                              hipStream_t stream)
{
  (void)in_sizes; (void)n_in; (void)out_size; (void)ws_size;
  const float* x1   = (const float*)d_in[0];
  const float* d2v  = (const float*)d_in[1];
  const float* Wih  = (const float*)d_in[2];
  const float* Whh  = (const float*)d_in[3];
  const float* bih  = (const float*)d_in[4];
  const float* bhh  = (const float*)d_in[5];
  const float* aW1  = (const float*)d_in[6];
  const float* ab1  = (const float*)d_in[7];
  const float* aw2  = (const float*)d_in[8];
  const float* rW   = (const float*)d_in[9];
  const float* rb   = (const float*)d_in[10];
  const float* ut   = (const float*)d_in[11];
  const float* at   = (const float*)d_in[12];
  const float* eW1  = (const float*)d_in[13];
  const float* eb1  = (const float*)d_in[14];
  const float* eW2  = (const float*)d_in[15];
  const float* eb2  = (const float*)d_in[16];
  const float* lW1  = (const float*)d_in[17];
  const float* lb1  = (const float*)d_in[18];
  const float* lW2  = (const float*)d_in[19];
  const float* lb2  = (const float*)d_in[20];
  const float* lW3  = (const float*)d_in[21];
  const float* lb3  = (const float*)d_in[22];
  const float* lW4  = (const float*)d_in[23];
  const float* lb4  = (const float*)d_in[24];
  const int* seq_lens   = (const int*)d_in[27];
  const int* user_ids   = (const int*)d_in[28];
  const int* sample_ids = (const int*)d_in[29];

  char* ws = (char*)d_ws;
  size_t off = 0;
  auto alloc = [&](size_t bytes)->char* {
    char* p = ws + off; off = (off + bytes + 255) & ~(size_t)255; return p;
  };
  _Float16* wihT = (_Float16*)alloc(4ull*512*1024*2);   // [l*2+d][512][1024] k-major
  _Float16* whhT = (_Float16*)alloc(4ull*256*1024*2);   // [l*2+d][256][1024] k-major
  _Float16* aW1T = (_Float16*)alloc(512ull*512*2);
  _Float16* rWT  = (_Float16*)alloc(512ull*64*2);
  _Float16* eW1T = (_Float16*)alloc(128ull*128*2);
  _Float16* eW2T = (_Float16*)alloc(128ull*64*2);
  _Float16* lW1T = (_Float16*)alloc(128ull*256*2);
  _Float16* lW2T = (_Float16*)alloc(256ull*128*2);
  _Float16* lW3T = (_Float16*)alloc(128ull*64*2);
  float*    bsum = (float*)alloc(4096ull*4);            // bih + bhh folded
  _Float16* x16  = (_Float16*)alloc((size_t)BT*512*2);  // f16 activations (reused)
  float*    xg   = (float*)alloc(2ull*TDIM*BDIM*G4*4);  // [d][t][b][1024]
  float*    rnn0 = (float*)alloc((size_t)BT*512*4);
  float*    rnn1 = (float*)alloc((size_t)BT*512*4);
  float*    ttmp = (float*)alloc((size_t)BT*512*4);
  float*    scb  = (float*)alloc((size_t)BT*4);
  float*    ctx  = (float*)alloc((size_t)BT*512*4);
  _Float16* z0   = (_Float16*)alloc((size_t)ROWS*128*2);
  _Float16* z1   = (_Float16*)alloc((size_t)ROWS*256*2);
  _Float16* z2   = (_Float16*)alloc((size_t)ROWS*128*2);
  _Float16* z3   = (_Float16*)alloc((size_t)ROWS*64*2);

  const dim3 blk(256);
  // ---- weight conversion (f32 -> f16, transpose to k-major) ----
  for (int i=0; i<4; ++i){
    k_convT<<<(1024*512+255)/256, blk, 0, stream>>>(Wih + (size_t)i*1024*512,
                                                    wihT + (size_t)i*512*1024, 1024, 512);
    k_convT<<<(1024*256+255)/256, blk, 0, stream>>>(Whh + (size_t)i*1024*256,
                                                    whhT + (size_t)i*256*1024, 1024, 256);
  }
  k_convT<<<(512*512+255)/256, blk, 0, stream>>>(aW1, aW1T, 512, 512);
  k_convT<<<(64*512+255)/256,  blk, 0, stream>>>(rW,  rWT,  64, 512);
  k_convT<<<(128*128+255)/256, blk, 0, stream>>>(eW1, eW1T, 128, 128);
  k_convT<<<(64*128+255)/256,  blk, 0, stream>>>(eW2, eW2T, 64, 128);
  k_convT<<<(256*128+255)/256, blk, 0, stream>>>(lW1, lW1T, 256, 128);
  k_convT<<<(128*256+255)/256, blk, 0, stream>>>(lW2, lW2T, 128, 256);
  k_convT<<<(64*128+255)/256,  blk, 0, stream>>>(lW3, lW3T, 64, 128);
  k_bias_sum<<<16, blk, 0, stream>>>(bih, bhh, bsum, 4096);

  // ---- bi-LSTM layer 0 ----
  k_cvt<<<(unsigned)(((size_t)BT*512+255)/256), blk, 0, stream>>>(x1, x16, (size_t)BT*512);
  for (int d=0; d<2; ++d)
    k_gemm<2><<<dim3(1024/64, BT/128), blk, 0, stream>>>(
        x16, wihT + (size_t)d*512*1024, bsum + d*1024,
        xg + (size_t)d*TDIM*BDIM*G4, BT, 1024, 512, 0);
  k_lstm<<<dim3(BDIM/16, 2), blk, 0, stream>>>(xg, whhT, seq_lens, rnn0);

  // ---- bi-LSTM layer 1 ----
  k_cvt<<<(unsigned)(((size_t)BT*512+255)/256), blk, 0, stream>>>(rnn0, x16, (size_t)BT*512);
  for (int d=0; d<2; ++d)
    k_gemm<2><<<dim3(1024/64, BT/128), blk, 0, stream>>>(
        x16, wihT + (size_t)(2+d)*512*1024, bsum + (2+d)*1024,
        xg + (size_t)d*TDIM*BDIM*G4, BT, 1024, 512, 0);
  k_lstm<<<dim3(BDIM/16, 2), blk, 0, stream>>>(xg, whhT + 2ull*256*1024, seq_lens, rnn1);

  // ---- attention ----
  k_cvt<<<(unsigned)(((size_t)BT*512+255)/256), blk, 0, stream>>>(rnn1, x16, (size_t)BT*512);
  k_gemm<1><<<dim3(512/64, BT/128), blk, 0, stream>>>(x16, aW1T, ab1, ttmp, BT, 512, 512, 2);
  k_attn_reduce<<<BT/256, blk, 0, stream>>>(ttmp, aw2, scb);
  k_ctx<<<BDIM, blk, 0, stream>>>(scb, x16, seq_lens, ctx);

  // ---- embedding + rnn branches -> z0 [ROWS][128] ----
  k_ebranch<<<ROWS/64, 128, 0, stream>>>(ut, at, user_ids, sample_ids,
                                         eW1T, eb1, eW2T, eb2, z0);
  k_rbranch<<<ROWS/128, blk, 0, stream>>>(ctx, d2v, rWT, rb, z0);

  // ---- last-MLP stack ----
  k_gemm<0><<<dim3(256/64, ROWS/128), blk, 0, stream>>>(z0, lW1T, lb1, z1, ROWS, 256, 128, 1);
  k_gemm<0><<<dim3(128/64, ROWS/128), blk, 0, stream>>>(z1, lW2T, lb2, z2, ROWS, 128, 256, 1);
  k_gemm<0><<<dim3(64/64,  ROWS/128), blk, 0, stream>>>(z2, lW3T, lb3, z3, ROWS, 64, 128, 1);
  k_final<<<ROWS/256, blk, 0, stream>>>(z3, lW4, lb4, (float*)d_out);
}
